// TestBinary_90434831385367
// MI455X (gfx1250) — matrix operand order repaired and checked by reading, not compiled, not run on hardware
//
#include <hip/hip_runtime.h>

typedef __attribute__((ext_vector_type(16))) _Float16 v16h;
typedef __attribute__((ext_vector_type(8)))  _Float16 v8h;
typedef __attribute__((ext_vector_type(8)))  float    v8f;

#define GEMM_M 8192
#define GEMM_N 4096
#define GEMM_K 4096

#define BM 128
#define BN 128
#define BK 32
#define LDT 40   // LDS row stride in halves: BK + 8 pad (keeps 16B alignment, spreads banks)

// ---------- pass 1: x fp32 -> f16 (row-major M x K) ----------
__global__ void __launch_bounds__(256) cvt_x_f16(const float* __restrict__ x,
                                                 _Float16* __restrict__ xh) {
  size_t i = ((size_t)blockIdx.x * 256 + threadIdx.x) * 8;
  float4 a = *(const float4*)(x + i);
  float4 b = *(const float4*)(x + i + 4);
  v8h h;
  h[0] = (_Float16)a.x; h[1] = (_Float16)a.y; h[2] = (_Float16)a.z; h[3] = (_Float16)a.w;
  h[4] = (_Float16)b.x; h[5] = (_Float16)b.y; h[6] = (_Float16)b.z; h[7] = (_Float16)b.w;
  *(v8h*)(xh + i) = h;
}

// ---------- pass 2: sign(kernel) fp32 (K x N) -> f16 +-1 transposed (N x K) ----------
__global__ void __launch_bounds__(256) sign_t_f16(const float* __restrict__ kr,
                                                  _Float16* __restrict__ bt) {
  __shared__ _Float16 tile[32][33];
  const int n  = blockIdx.x * 32 + threadIdx.x;
  const int k0 = blockIdx.y * 32;
#pragma unroll
  for (int i = 0; i < 32; i += 8) {
    float v = kr[(size_t)(k0 + threadIdx.y + i) * GEMM_N + n];
    tile[threadIdx.y + i][threadIdx.x] = (v >= 0.0f) ? (_Float16)1.0f : (_Float16)-1.0f;
  }
  __syncthreads();
  const int nb = blockIdx.x * 32;
#pragma unroll
  for (int i = 0; i < 32; i += 8) {
    bt[(size_t)(nb + threadIdx.y + i) * GEMM_K + k0 + threadIdx.x] =
        tile[threadIdx.x][threadIdx.y + i];
  }
}

// ---------- pass 3: C(f32, MxN) = A(f16, MxK) * Bt(f16 +-1, NxK)^T via WMMA ----------
__global__ void __launch_bounds__(256) bsign_gemm(const _Float16* __restrict__ A,
                                                  const _Float16* __restrict__ Bt,
                                                  float* __restrict__ C) {
  __shared__ __align__(16) _Float16 sA[2][BM * LDT];
  __shared__ __align__(16) _Float16 sB[2][BN * LDT];

  const int tid  = threadIdx.x;
  const int lane = tid & 31;
  const int wave = tid >> 5;
  const int wm   = (wave & 1) * 64;   // wave row offset inside block tile
  const int wn   = (wave >> 1) * 32;  // wave col offset inside block tile

  const size_t bm = (size_t)blockIdx.y * BM;
  const size_t bn = (size_t)blockIdx.x * BN;

  // global->LDS loader mapping: 2 threads per row, 16 halves (32B) each
  const int lrow = tid >> 1;
  const int lseg = (tid & 1) * 16;
  const int sidx = lrow * LDT + lseg;

  const _Float16* gA = A  + (bm + lrow) * (size_t)GEMM_K + lseg;
  const _Float16* gB = Bt + (bn + lrow) * (size_t)GEMM_K + lseg;

  // fragment lane decomposition (per ISA 16-bit A/B VGPR layouts, wave32)
  const int fr  = lane & 15;          // row (A) / col (B) within 16x16 tile
  const int akb = (lane >> 4) * 8;    // A: lanes 16-31 hold K+8 block
  const int bkb = (lane >> 4) * 8;    // B: lanes 16-31 hold K+8 block

  v8f acc[4][2] = {};

  auto load_tile = [&](int buf, size_t koff) {
    uint4 a0 = *(const uint4*)(gA + koff);
    uint4 a1 = *(const uint4*)(gA + koff + 8);
    uint4 b0 = *(const uint4*)(gB + koff);
    uint4 b1 = *(const uint4*)(gB + koff + 8);
    *(uint4*)&sA[buf][sidx]     = a0;
    *(uint4*)&sA[buf][sidx + 8] = a1;
    *(uint4*)&sB[buf][sidx]     = b0;
    *(uint4*)&sB[buf][sidx + 8] = b1;
  };

  load_tile(0, 0);
  __syncthreads();

  const int nk = GEMM_K / BK;
  for (int kt = 0; kt < nk; ++kt) {
    const int buf = kt & 1;
    if (kt + 1 < nk) load_tile(buf ^ 1, (size_t)(kt + 1) * BK);

    const _Float16* sa = &sA[buf][0];
    const _Float16* sb = &sB[buf][0];

    v16h af[4], bfr[2];
#pragma unroll
    for (int i = 0; i < 4; ++i) {
      const int row = wm + i * 16 + fr;
      union { v16h v; v8h h[2]; } u;
      u.h[0] = *(const v8h*)&sa[row * LDT + akb];
      u.h[1] = *(const v8h*)&sa[row * LDT + akb + 16];
      af[i] = u.v;
    }
#pragma unroll
    for (int j = 0; j < 2; ++j) {
      const int col = wn + j * 16 + fr;
      union { v16h v; v8h h[2]; } u;
      u.h[0] = *(const v8h*)&sb[col * LDT + bkb];
      u.h[1] = *(const v8h*)&sb[col * LDT + bkb + 16];
      bfr[j] = u.v;
    }

#pragma unroll
    for (int i = 0; i < 4; ++i)
#pragma unroll
      for (int j = 0; j < 2; ++j)
        acc[i][j] = __builtin_amdgcn_wmma_f32_16x16x32_f16(
            false, af[i], false, bfr[j], (short)0, acc[i][j], false, false);

    __syncthreads();
  }

  // epilogue: C/D layout — VGPR r: lanes 0-15 -> M=r, lanes 16-31 -> M=r+8
#pragma unroll
  for (int i = 0; i < 4; ++i) {
    const size_t row0 = bm + wm + i * 16 + (lane >> 4) * 8;
#pragma unroll
    for (int j = 0; j < 2; ++j) {
      const size_t col = bn + wn + j * 16 + fr;
#pragma unroll
      for (int r = 0; r < 8; ++r)
        C[(row0 + r) * (size_t)GEMM_N + col] = acc[i][j][r];
    }
  }
}

extern "C" void kernel_launch(void* const* d_in, const int* in_sizes, int n_in,
                              void* d_out, int out_size, void* d_ws, size_t ws_size,
                              hipStream_t stream) {
  (void)in_sizes; (void)n_in; (void)out_size; (void)ws_size;
  const float* x  = (const float*)d_in[0];
  const float* kr = (const float*)d_in[1];
  float* out = (float*)d_out;

  _Float16* xh = (_Float16*)d_ws;                      // M*K f16   (64 MiB)
  _Float16* bt = xh + (size_t)GEMM_M * GEMM_K;         // N*K f16   (32 MiB)

  const size_t nx = (size_t)GEMM_M * GEMM_K;
  cvt_x_f16<<<dim3((unsigned)(nx / (256 * 8))), dim3(256), 0, stream>>>(x, xh);
  sign_t_f16<<<dim3(GEMM_N / 32, GEMM_K / 32), dim3(32, 8), 0, stream>>>(kr, bt);
  bsign_gemm<<<dim3(GEMM_N / BN, GEMM_M / BM), dim3(256), 0, stream>>>(xh, bt, out);
}
